// GPS__16466904612871
// MI455X (gfx1250) — matrix operand order repaired and checked by reading, not compiled, not run on hardware
//
#include <hip/hip_runtime.h>
#include <hip/hip_bf16.h>

// GCN forward on MI455X (gfx1250, wave32).
// Memory-bound: ~1.3 GB of edge gather/scatter dominates (≈57us at 23.3 TB/s).
// GEMMs (3.3 GFLOP) run on WMMA bf16->f32; packed W staged to LDS via the
// Tensor Data Mover (tensor_load_to_lds + s_wait_tensorcnt), B fragments
// batch-loaded from LDS so WMMAs issue back-to-back.

typedef __attribute__((ext_vector_type(16))) __bf16 v16bf;
typedef __attribute__((ext_vector_type(8)))  __bf16 v8bf;
typedef __attribute__((ext_vector_type(8)))  float  v8f;
typedef __attribute__((ext_vector_type(4)))  unsigned int v4u;
typedef __attribute__((ext_vector_type(8)))  int v8i;
typedef __attribute__((ext_vector_type(4)))  int v4i;

#define HD 128   // feature width (D == H == 128)
#define GC 64    // graphs
#define NC 10    // classes

// ---------------- small utility kernels ----------------

__global__ void k_fill_f32(float* p, float v, int n) {
    int i = blockIdx.x * blockDim.x + threadIdx.x;
    if (i < n) p[i] = v;
}

__global__ void k_deg_accum(const int* col, float* deg, int e) {
    int i = blockIdx.x * blockDim.x + threadIdx.x;
    if (i < e) atomicAdd(&deg[col[i]], 1.0f);
}

__global__ void k_rsqrt_inplace(float* d, int n) {
    int i = blockIdx.x * blockDim.x + threadIdx.x;
    if (i < n) d[i] = __frsqrt_rn(d[i]);
}

__global__ void k_cvt_bf16(const float* __restrict__ src, __bf16* __restrict__ dst, int n) {
    int i = blockIdx.x * blockDim.x + threadIdx.x;
    if (i < n) dst[i] = (__bf16)src[i];
}

__global__ void k_relu_cvt_bf16(const float* __restrict__ src, __bf16* __restrict__ dst, int n) {
    int i = blockIdx.x * blockDim.x + threadIdx.x;
    if (i < n) dst[i] = (__bf16)fmaxf(src[i], 0.0f);
}

// Pack W[128][128] (K x N, row-major) into per-lane WMMA B fragments (bf16).
// Fragment (ct,kt): lane L holds 16 bf16 contiguous: n = ct*16 + (L&15),
// k = kt*32 + (L>=16 ? 16 : 0) + j  (j = 0..15).
__global__ void k_pack_w(const float* __restrict__ W, __bf16* __restrict__ Wp) {
    int tid = blockIdx.x * blockDim.x + threadIdx.x;  // 0..16383
    if (tid >= HD * HD) return;
    int fragid = tid >> 9;        // 32 frags of 512 elements
    int r      = tid & 511;
    int L      = r >> 4;
    int j      = r & 15;
    int ct     = fragid >> 2;
    int kt     = fragid & 3;
    int k      = kt * 32 + ((L >> 4) << 4) + j;
    int n      = ct * 16 + (L & 15);
    Wp[tid] = (__bf16)W[k * HD + n];
}

// ---------------- TDM helper: async-copy 32KB packed W into LDS ------------
// Builds a 2-D tensor descriptor (D#): bf16 elements (data_size=1 -> 2B),
// tensor_dim0 = tile_dim0 = 16384 elements, one row.  TENSORcnt-tracked.
__device__ __forceinline__ void tdm_load_wp(const __bf16* Wp, unsigned ldsOff) {
    unsigned long long ga = (unsigned long long)(uintptr_t)Wp;

    v4u g0;
    g0[0] = 1u;                                           // count=1, user mode
    g0[1] = ldsOff;                                       // lds_addr (bytes)
    g0[2] = (unsigned)(ga & 0xffffffffu);                 // global_addr[31:0]
    g0[3] = (unsigned)((ga >> 32) & 0x01ffffffu)          // global_addr[56:32]
          | (2u << 30);                                   // type = 2 (image)

    const unsigned elems = (unsigned)(HD * HD);           // 16384 bf16
    v8i g1;
    g1[0] = (int)(1u << 16);                              // wg_mask=0, data_size=1 (2B)
    g1[1] = (int)((elems & 0xffffu) << 16);               // tensor_dim0[15:0]
    g1[2] = (int)((elems >> 16) | (1u << 16));            // dim0[31:16] | tensor_dim1=1
    g1[3] = (int)((elems & 0xffffu) << 16);               // tile_dim0 = 16384
    g1[4] = 1;                                            // tile_dim1=1, tile_dim2=0
    g1[5] = (int)elems;                                   // tensor_dim0_stride[31:0]
    g1[6] = 0;                                            // stride0[47:32] | stride1[15:0]
    g1[7] = 0;

    v4i gz = {0, 0, 0, 0};
#if defined(__clang_major__) && (__clang_major__ >= 23)
    v8i gz8 = {0, 0, 0, 0, 0, 0, 0, 0};
    __builtin_amdgcn_tensor_load_to_lds(g0, g1, gz, gz, gz8, 0);
#else
    __builtin_amdgcn_tensor_load_to_lds(g0, g1, gz, gz, 0);
#endif
    __builtin_amdgcn_s_wait_tensorcnt(0);                 // TDM done for this wave
}

// ---------------- WMMA GEMM: H[n][128] = Xb[n][128] @ W (packed Wp) --------
// One wave per 16-row tile; each wave computes all 8 column tiles.
// 8 waves per block -> 128 rows per block.  N is a multiple of 16 (50000).
__global__ __launch_bounds__(256) void k_gemm_bf16(const __bf16* __restrict__ Xb,
                                                   const __bf16* __restrict__ Wp,
                                                   float* __restrict__ Hout,
                                                   int rowTiles) {
    __shared__ __bf16 lWp[HD * HD];                       // 32 KB packed W

    if (threadIdx.x == 0) {
        unsigned ldsOff = (unsigned)(unsigned long long)(uintptr_t)(&lWp[0]);
        tdm_load_wp(Wp, ldsOff);                          // TDM + wait tensorcnt
    }
    __syncthreads();                                      // LDS visible to all waves

    const int wave = threadIdx.x >> 5;
    const int lane = threadIdx.x & 31;
    const int rt   = blockIdx.x * 8 + wave;

    if (rt < rowTiles) {       // wave-uniform: EXEC all-1s inside (WMMA req.)
        const int m  = lane & 15;                         // A row / C-D column
        const int hi = lane >> 4;                         // K-half selector
        const long rowbase = (long)(rt * 16 + m) * HD;

        v8f acc[8];
#pragma unroll
        for (int ct = 0; ct < 8; ++ct) acc[ct] = (v8f)0.0f;

#pragma unroll
        for (int kt = 0; kt < 4; ++kt) {
            // A fragment: 16-bit 16x32 layout -> two contiguous 16B chunks
            v8bf a0 = *(const v8bf*)(Xb + rowbase + kt * 32 + hi * 16);
            v8bf a1 = *(const v8bf*)(Xb + rowbase + kt * 32 + hi * 16 + 8);
            v16bf a;
#pragma unroll
            for (int j = 0; j < 8; ++j) { a[j] = a0[j]; a[j + 8] = a1[j]; }

            if (kt < 3)   // prefetch next K-slice of A (global_prefetch_b8)
                __builtin_prefetch((const void*)(Xb + rowbase + (kt + 1) * 32), 0, 1);

            // Batch-load all 8 B fragments of this K-slice into distinct
            // registers (one ds_load clause), then issue 8 WMMAs back-to-back.
            v16bf bf[8];
#pragma unroll
            for (int ct = 0; ct < 8; ++ct)
                bf[ct] = *(const v16bf*)(lWp + (((ct << 2) + kt) * 32 + lane) * 16);

#pragma unroll
            for (int ct = 0; ct < 8; ++ct)
                acc[ct] = __builtin_amdgcn_wmma_f32_16x16x32_bf16(
                    false, a, false, bf[ct], (short)0, acc[ct], false, false);
        }

        // C/D layout: VGPR r of lane L -> row (r + hi*8), col (L&15)
#pragma unroll
        for (int ct = 0; ct < 8; ++ct) {
#pragma unroll
            for (int r = 0; r < 8; ++r) {
                int mm = r + hi * 8;
                Hout[(long)(rt * 16 + mm) * HD + ct * 16 + m] = acc[ct][r];
            }
        }
    }
}

// ---------------- aggregation kernels ----------------

// out[i][f] = bias[f] + dinv[i]^2 * H[i][f]   (self-loop message + bias)
__global__ void k_agg_init(const float* __restrict__ H, const float* __restrict__ dinv,
                           const float* __restrict__ bias, float* __restrict__ out, int n) {
    int idx = blockIdx.x * blockDim.x + threadIdx.x;
    if (idx >= n * HD) return;
    int node = idx >> 7;
    int f    = idx & (HD - 1);
    float s  = dinv[node];
    out[idx] = bias[f] + s * s * H[idx];
}

// out[col] += dinv[row]*dinv[col] * H[row]; one wave per edge, 4 floats/lane
__global__ void k_agg_edges(const int* __restrict__ row, const int* __restrict__ col,
                            const float* __restrict__ dinv, const float* __restrict__ H,
                            float* __restrict__ out, int e) {
    int gid  = blockIdx.x * blockDim.x + threadIdx.x;
    int edge = gid >> 5;
    if (edge >= e) return;
    int lane = gid & 31;
    int r = row[edge];
    int c = col[edge];
    float nrm = dinv[r] * dinv[c];
    const float4 v = *(const float4*)(H + (long)r * HD + lane * 4);
    float* dst = out + (long)c * HD + lane * 4;
    atomicAdd(dst + 0, nrm * v.x);
    atomicAdd(dst + 1, nrm * v.y);
    atomicAdd(dst + 2, nrm * v.z);
    atomicAdd(dst + 3, nrm * v.w);
}

// ---------------- pooling + head ----------------

__global__ void k_pool_accum(const int* __restrict__ batch, const float* __restrict__ h,
                             float* __restrict__ sums, float* __restrict__ cnt, int n) {
    int gid  = blockIdx.x * blockDim.x + threadIdx.x;
    int node = gid >> 5;
    if (node >= n) return;
    int lane = gid & 31;
    int g = batch[node];
    const float4 v = *(const float4*)(h + (long)node * HD + lane * 4);
    float* dst = sums + (long)g * HD + lane * 4;
    atomicAdd(dst + 0, v.x);
    atomicAdd(dst + 1, v.y);
    atomicAdd(dst + 2, v.z);
    atomicAdd(dst + 3, v.w);
    if (lane == 0) atomicAdd(&cnt[g], 1.0f);
}

__global__ void k_head(const float* __restrict__ sums, const float* __restrict__ cnt,
                       const float* __restrict__ Wlin, const float* __restrict__ blin,
                       float* __restrict__ out) {
    int idx = blockIdx.x * blockDim.x + threadIdx.x;
    if (idx >= GC * NC) return;
    int g = idx / NC;
    int c = idx % NC;
    float inv = 1.0f / fmaxf(cnt[g], 1.0f);
    float acc = blin[c];
#pragma unroll 8
    for (int f = 0; f < HD; ++f)
        acc += sums[g * HD + f] * inv * Wlin[f * NC + c];
    out[idx] = acc;
}

// ---------------- launcher ----------------

static inline size_t alignup(size_t x) { return (x + 255) & ~(size_t)255; }

extern "C" void kernel_launch(void* const* d_in, const int* in_sizes, int n_in,
                              void* d_out, int out_size, void* d_ws, size_t ws_size,
                              hipStream_t stream) {
    const float* x    = (const float*)d_in[0];
    const float* W1   = (const float*)d_in[1];
    const float* b1   = (const float*)d_in[2];
    const float* W2   = (const float*)d_in[3];
    const float* b2   = (const float*)d_in[4];
    const float* Wlin = (const float*)d_in[5];
    const float* blin = (const float*)d_in[6];
    const int*   ei   = (const int*)d_in[7];
    const int*   batch = (const int*)d_in[8];
    float* out = (float*)d_out;

    const int N = in_sizes[8];            // 50000 nodes
    const int E = in_sizes[7] / 2;        // 600000 edges
    const int rowTiles = N / 16;          // 3125 (N multiple of 16)
    const int*  row = ei;
    const int*  col = ei + E;

    // workspace carve-up
    char* ws = (char*)d_ws;
    size_t off = 0;
    float*  dinv  = (float*) (ws + off); off += alignup((size_t)N * 4);
    __bf16* Xb    = (__bf16*)(ws + off); off += alignup((size_t)N * HD * 2);
    __bf16* Wp1   = (__bf16*)(ws + off); off += alignup((size_t)HD * HD * 2);
    __bf16* Wp2   = (__bf16*)(ws + off); off += alignup((size_t)HD * HD * 2);
    float*  Hbuf  = (float*) (ws + off); off += alignup((size_t)N * HD * 4);
    float*  Abuf  = (float*) (ws + off); off += alignup((size_t)N * HD * 4);
    float*  sums  = (float*) (ws + off); off += alignup((size_t)GC * HD * 4);
    float*  cnt   = (float*) (ws + off); off += alignup((size_t)GC * 4);
    (void)ws_size; (void)n_in; (void)out_size;

    const int T = 256;
    const int nf = N * HD;

    // degrees (self-loop => init 1.0), then dinv = rsqrt(deg)
    k_fill_f32<<<(N + T - 1) / T, T, 0, stream>>>(dinv, 1.0f, N);
    k_deg_accum<<<(E + T - 1) / T, T, 0, stream>>>(col, dinv, E);
    k_rsqrt_inplace<<<(N + T - 1) / T, T, 0, stream>>>(dinv, N);

    // bf16 operand prep
    k_cvt_bf16<<<(nf + T - 1) / T, T, 0, stream>>>(x, Xb, nf);
    k_pack_w<<<(HD * HD + T - 1) / T, T, 0, stream>>>(W1, Wp1);
    k_pack_w<<<(HD * HD + T - 1) / T, T, 0, stream>>>(W2, Wp2);

    const int gemmBlocks = (rowTiles + 7) / 8;

    // ---- layer 1 ----
    k_gemm_bf16<<<gemmBlocks, 256, 0, stream>>>(Xb, Wp1, Hbuf, rowTiles);
    k_agg_init<<<(nf + T - 1) / T, T, 0, stream>>>(Hbuf, dinv, b1, Abuf, N);
    k_agg_edges<<<((long)E * 32 + T - 1) / T, T, 0, stream>>>(row, col, dinv, Hbuf, Abuf, E);
    k_relu_cvt_bf16<<<(nf + T - 1) / T, T, 0, stream>>>(Abuf, Xb, nf);

    // ---- layer 2 ----
    k_gemm_bf16<<<gemmBlocks, 256, 0, stream>>>(Xb, Wp2, Hbuf, rowTiles);
    k_agg_init<<<(nf + T - 1) / T, T, 0, stream>>>(Hbuf, dinv, b2, Abuf, N);
    k_agg_edges<<<((long)E * 32 + T - 1) / T, T, 0, stream>>>(row, col, dinv, Hbuf, Abuf, E);

    // ---- mean pool + classifier head ----
    k_fill_f32<<<(GC * HD + T - 1) / T, T, 0, stream>>>(sums, 0.0f, GC * HD);
    k_fill_f32<<<(GC + T - 1) / T, T, 0, stream>>>(cnt, 0.0f, GC);
    k_pool_accum<<<((long)N * 32 + T - 1) / T, T, 0, stream>>>(batch, Abuf, sums, cnt, N);
    k_head<<<(GC * NC + T - 1) / T, T, 0, stream>>>(sums, cnt, Wlin, blin, out);
}
